// SNN_32512902431335
// MI455X (gfx1250) — matrix operand order repaired and checked by reading, not compiled, not run on hardware
//
#include <hip/hip_runtime.h>

#define B_SAMPLES 32768
#define IN_DIM    784
#define HID       128
#define OUT_DIM   10
#define TSTEPS    20
#define DECAYF    0.25f
#define THRESHF   1.0f

#define MROWS     64           // rows (samples) per block
#define MTILES    (MROWS/16)   // 4 accumulator tiles per wave

typedef __attribute__((ext_vector_type(16))) __bf16 v16bf;
typedef __attribute__((ext_vector_type(8)))  __bf16 v8bf;
typedef __attribute__((ext_vector_type(4)))  __bf16 v4bf;
typedef __attribute__((ext_vector_type(8)))  float  v8f;
typedef __attribute__((ext_vector_type(4)))  float  v4f;

// split one v4f into bf16 hi + bf16 lo residual (RNE both ways)
__device__ __forceinline__ void cvt_split4(const v4f v, v4bf& h, v4bf& l) {
  #pragma unroll
  for (int j = 0; j < 4; ++j) {
    const float f = v[j];
    const __bf16 hi = (__bf16)f;
    h[j] = hi;
    l[j] = (__bf16)(f - (float)hi);
  }
}

// LDS byte address of a generic pointer to __shared__ (aperture low 32 bits)
__device__ __forceinline__ unsigned lds_addr(const void* p) {
  return (unsigned)(size_t)p;
}

// async copy 16 bytes global -> LDS (GV mode), tracked by ASYNCcnt
__device__ __forceinline__ void async_copy_b128(unsigned lds_byte_addr,
                                                const void* gptr) {
  asm volatile("global_load_async_to_lds_b128 %0, %1, off"
               :: "v"(lds_byte_addr),
                  "v"((unsigned long long)(size_t)gptr)
               : "memory");
}

__device__ __forceinline__ void wait_asynccnt0() {
  asm volatile("s_wait_asynccnt 0x0" ::: "memory");
}

// ---------------------------------------------------------------------------
// Kernel 1: cur1[B,128] = x[B,784] @ W1[128,784]^T + b1[128]
// Split-precision bf16 WMMA (acc += xh*wh + xh*wl + xl*wh, f32 accumulate).
// Block = 256 threads = 8 waves; block tile = 64 rows x 128 cols.
// A (x) tile: sync-staged as split-bf16 (shared by all 8 waves).
// B (W1) tile: async global->LDS copy of raw f32 (ASYNCcnt); f32->bf16 split
//   happens at fragment build (each element consumed by exactly one lane).
// K loop: 24 guard-free 32-wide chunks + one zero-padded 16-wide tail.
// ---------------------------------------------------------------------------
__global__ __launch_bounds__(256) void snn_gemm1(const float* __restrict__ x,
                                                 const float* __restrict__ W1,
                                                 const float* __restrict__ b1,
                                                 float* __restrict__ cur1) {
  __shared__ __align__(64) __bf16 sAh[MROWS][32];  // 4 KB
  __shared__ __align__(64) __bf16 sAl[MROWS][32];  // 4 KB
  __shared__ __align__(64) float  sBf[HID][32];    // 16 KB, async-filled f32

  const int tid  = threadIdx.x;
  const int wave = tid >> 5;
  const int lane = tid & 31;
  const int half = lane >> 4;   // 0: lanes 0-15, 1: lanes 16-31
  const int l16  = lane & 15;
  const int row0 = blockIdx.x * MROWS;
  const int col0 = wave * 16;

  v8f acc[MTILES] = {};

  for (int chunk = 0; chunk < 25; ++chunk) {
    const int k0 = chunk * 32;
    if (chunk < 24) {                       // uniform branch: EXEC stays full
      // B (W1): 128 n x 32 k = 1024 b128 slots, 4 async copies per thread
      #pragma unroll
      for (int it = 0; it < 4; ++it) {
        const int idx = tid + it * 256;
        const int n = idx >> 3, kq = (idx & 7) * 4;
        async_copy_b128(lds_addr(&sBf[n][kq]),
                        &W1[(size_t)n * IN_DIM + k0 + kq]);
      }
      // A (x): 64 rows x 32 k = 512 v4f slots, 2 per thread (sync + convert)
      #pragma unroll
      for (int it = 0; it < 2; ++it) {
        const int idx = tid + it * 256;
        const int r = idx >> 3, kq = (idx & 7) * 4;
        v4f v = __builtin_nontemporal_load(
            (const v4f*)&x[(size_t)(row0 + r) * IN_DIM + k0 + kq]);
        v4bf h, l; cvt_split4(v, h, l);
        *(v4bf*)&sAh[r][kq] = h;  *(v4bf*)&sAl[r][kq] = l;
      }
    } else {
      // Tail chunk: k0 = 768, 16 valid columns, zero-pad k = 16..31.
      #pragma unroll
      for (int it = 0; it < 2; ++it) {  // B valid: 128 x 16 = 512 slots
        const int idx = tid + it * 256;
        const int n = idx >> 2, kq = (idx & 3) * 4;
        async_copy_b128(lds_addr(&sBf[n][kq]),
                        &W1[(size_t)n * IN_DIM + k0 + kq]);
        const v4f z = {0.f, 0.f, 0.f, 0.f};
        *(v4f*)&sBf[n][16 + kq] = z;       // pad (ds store, barrier-ordered)
      }
      {  // A valid: 64 x 16 = 256 v4f slots, 1 per thread
        const int r = tid >> 2, kq = (tid & 3) * 4;
        v4f v = __builtin_nontemporal_load(
            (const v4f*)&x[(size_t)(row0 + r) * IN_DIM + k0 + kq]);
        v4bf h, l; cvt_split4(v, h, l);
        *(v4bf*)&sAh[r][kq] = h;  *(v4bf*)&sAl[r][kq] = l;
        const v4bf z = {};
        *(v4bf*)&sAh[r][16 + kq] = z;  *(v4bf*)&sAl[r][16 + kq] = z;
      }
    }
    wait_asynccnt0();       // all of this wave's async copies landed in LDS
    __syncthreads();        // cross-wave visibility

    // B fragment (32x16 bf16): lane N = l16; K = half*16 + j (16 contiguous).
    // Read raw f32 from LDS, split to bf16 hi/lo in registers.
    const float* bp = &sBf[col0 + l16][half * 16];
    v16bf bh, bl;
    #pragma unroll
    for (int q = 0; q < 4; ++q) {
      const v4f bv = *(const v4f*)(bp + q * 4);
      v4bf h, l; cvt_split4(bv, h, l);
      #pragma unroll
      for (int e = 0; e < 4; ++e) { bh[q * 4 + e] = h[e]; bl[q * 4 + e] = l[e]; }
    }

    #pragma unroll
    for (int m = 0; m < MTILES; ++m) {
      // A fragment (16x32 bf16): lane M = l16;
      // j=0..15 -> K = half*16 + j (16 contiguous)
      const int ar = m * 16 + l16;
      v8bf a0h = *(const v8bf*)&sAh[ar][half * 16];
      v8bf a1h = *(const v8bf*)&sAh[ar][half * 16 + 8];
      v8bf a0l = *(const v8bf*)&sAl[ar][half * 16];
      v8bf a1l = *(const v8bf*)&sAl[ar][half * 16 + 8];
      v16bf ah, al;
      #pragma unroll
      for (int j = 0; j < 8; ++j) {
        ah[j] = a0h[j]; ah[8 + j] = a1h[j];
        al[j] = a0l[j]; al[8 + j] = a1l[j];
      }
      acc[m] = __builtin_amdgcn_wmma_f32_16x16x32_bf16(false, ah, false, bh, (short)0, acc[m], false, false);
      acc[m] = __builtin_amdgcn_wmma_f32_16x16x32_bf16(false, ah, false, bl, (short)0, acc[m], false, false);
      acc[m] = __builtin_amdgcn_wmma_f32_16x16x32_bf16(false, al, false, bh, (short)0, acc[m], false, false);
    }
    __syncthreads();
  }

  // C layout: VGPR i -> M = i + half*8, N = l16
  const int   col  = col0 + l16;
  const float bias = b1[col];
  #pragma unroll
  for (int m = 0; m < MTILES; ++m) {
    #pragma unroll
    for (int i = 0; i < 8; ++i) {
      const int row = row0 + m * 16 + half * 8 + i;
      cur1[(size_t)row * HID + col] = acc[m][i] + bias;
    }
  }
}

// ---------------------------------------------------------------------------
// Kernel 2: 20-step LIF dynamics. One wave32 per sample; lane owns 4 channels.
// spk1 stored per step (nontemporal v4f, coalesced). 128->10 GEMV via
// per-lane FMAs + wave32 xor-shuffle reductions; everything else in registers.
// ---------------------------------------------------------------------------
__global__ __launch_bounds__(256) void snn_dyn(const float* __restrict__ cur1,
                                               const float* __restrict__ W2,
                                               const float* __restrict__ b2,
                                               float* __restrict__ out,
                                               float* __restrict__ spk1_rec) {
  const int lane   = threadIdx.x & 31;
  const int wave   = threadIdx.x >> 5;
  const int sample = blockIdx.x * 8 + wave;
  const int c0     = lane * 4;

  const v4f cur = *(const v4f*)&cur1[(size_t)sample * HID + c0];

  v4f   w2r[OUT_DIM];
  float b2r[OUT_DIM];
  #pragma unroll
  for (int n = 0; n < OUT_DIM; ++n) {
    w2r[n] = *(const v4f*)&W2[(size_t)n * HID + c0];
    b2r[n] = b2[n];
  }

  v4f   mem1 = {0.f, 0.f, 0.f, 0.f};
  float mem2[OUT_DIM], osum[OUT_DIM];
  #pragma unroll
  for (int n = 0; n < OUT_DIM; ++n) { mem2[n] = 0.f; osum[n] = 0.f; }

  for (int t = 0; t < TSTEPS; ++t) {
    v4f spk;
    #pragma unroll
    for (int j = 0; j < 4; ++j) {
      mem1[j] = mem1[j] * DECAYF + cur[j];
      spk[j]  = (mem1[j] > THRESHF) ? 1.f : 0.f;
      mem1[j] *= (1.f - spk[j]);
    }

    __builtin_nontemporal_store(
        spk, (v4f*)&spk1_rec[((size_t)t * B_SAMPLES + sample) * HID + c0]);

    #pragma unroll
    for (int n = 0; n < OUT_DIM; ++n) {
      float p = spk[0] * w2r[n][0];
      p = fmaf(spk[1], w2r[n][1], p);
      p = fmaf(spk[2], w2r[n][2], p);
      p = fmaf(spk[3], w2r[n][3], p);
      #pragma unroll
      for (int off = 16; off > 0; off >>= 1)
        p += __shfl_xor(p, off, 32);          // replicated sum across wave32
      mem2[n] = mem2[n] * DECAYF + p + b2r[n];
      const float s2 = (mem2[n] > THRESHF) ? 1.f : 0.f;
      mem2[n] *= (1.f - s2);
      osum[n] += s2;
    }
  }

  if (lane == 0) {
    #pragma unroll
    for (int n = 0; n < OUT_DIM; ++n)
      out[(size_t)sample * OUT_DIM + n] = osum[n];
  }
}

// ---------------------------------------------------------------------------
extern "C" void kernel_launch(void* const* d_in, const int* in_sizes, int n_in,
                              void* d_out, int out_size, void* d_ws, size_t ws_size,
                              hipStream_t stream) {
  (void)in_sizes; (void)n_in; (void)out_size; (void)ws_size;
  const float* x  = (const float*)d_in[0];   // [B,784]
  const float* W1 = (const float*)d_in[1];   // [128,784]
  const float* b1 = (const float*)d_in[2];   // [128]
  const float* W2 = (const float*)d_in[3];   // [10,128]
  const float* b2 = (const float*)d_in[4];   // [10]

  float* out      = (float*)d_out;                              // [B,10]
  float* spk1_rec = out + (size_t)B_SAMPLES * OUT_DIM;          // [20,B,128]
  float* cur1     = (float*)d_ws;                               // [B,128] f32

  snn_gemm1<<<B_SAMPLES / MROWS, 256, 0, stream>>>(x, W1, b1, cur1);
  snn_dyn  <<<B_SAMPLES / 8,    256, 0, stream>>>(cur1, W2, b2, out, spk1_rec);
}
